// GroupQueryAttention_29583734735602
// MI455X (gfx1250) — matrix operand order repaired and checked by reading, not compiled, not run on hardware
//
#include <hip/hip_runtime.h>
#include <hip/hip_bf16.h>
#include <stdint.h>

// GQA block for MI455X (gfx1250, wave32, WMMA 16x16x32 f16 -> f32 acc).
// Pipeline: rmsnorm->f16, weight cast+transpose, WMMA GEMM (+RoPE epilogue,
// 32-row tiles, single base pointer per operand stream, launch_bounds(128,1)
// so the allocator may use ~200 VGPRs without spilling), flash attention
// (WMMA scores + WMMA P*V with LDS P exchange), WMMA out-proj.

typedef _Float16 half_t;
typedef __attribute__((ext_vector_type(16))) _Float16 v16h;
typedef __attribute__((ext_vector_type(8)))  float    v8f;

#define HIDDEN   2048
#define NHEADS   16
#define NKVHEADS 4
#define HEADDIM  128
#define SEQ      2048
#define BATCH    2
#define MTOT     (BATCH * SEQ)          // 4096 rows
#define KD       HIDDEN                 // contraction dim for all GEMMs

// ---------------------------------------------------------------------------
// Fragment loaders (contiguous k = 16*(lane/16) + s, wave32).  Pointers are pre-offset
// per lane (A: +row*ld + khalf ; B: +col*ld + kb) so every displacement
// below is a compile-time constant that folds into the load IOFFSET.
// ---------------------------------------------------------------------------

// A 16x32 f16: element pairs at +{0,2,...,14} halves from base.
static __device__ inline v16h load_a_frag_pre(const half_t* p) {
  v16h f;
#pragma unroll
  for (int v = 0; v < 8; ++v) {
    const int off = 2 * v;
    union { uint32_t u; half_t h[2]; } cv;
    cv.u = *(const uint32_t*)(p + off);
    f[2 * v]     = cv.h[0];
    f[2 * v + 1] = cv.h[1];
  }
  return f;
}

// B 32x16 f16: element pairs at +{0,2,...,14} halves from base.
static __device__ inline v16h load_b_frag_pre(const half_t* p) {
  v16h f;
#pragma unroll
  for (int v = 0; v < 8; ++v) {
    union { uint32_t u; half_t h[2]; } cv;
    cv.u = *(const uint32_t*)(p + 2 * v);
    f[2 * v]     = cv.h[0];
    f[2 * v + 1] = cv.h[1];
  }
  return f;
}

static __device__ inline v8f wmma_f16(v16h a, v16h b, v8f c) {
  return __builtin_amdgcn_wmma_f32_16x16x32_f16(false, a, false, b,
                                                (short)0, c, false, false);
}

// ---------------------------------------------------------------------------
// Kernel 1: RMSNorm + cast to f16.  One block per row of 2048.
// ---------------------------------------------------------------------------
__global__ void rmsnorm_cast(const float* __restrict__ x,
                             const float* __restrict__ w,
                             half_t* __restrict__ xn) {
  const int row = blockIdx.x;
  const float* xr = x + (size_t)row * HIDDEN;
  float ss = 0.f;
  for (int i = threadIdx.x; i < HIDDEN; i += 256) {
    float v = xr[i];
    ss += v * v;
  }
#pragma unroll
  for (int off = 16; off > 0; off >>= 1) ss += __shfl_xor(ss, off, 32);
  __shared__ float red[8];
  const int lane = threadIdx.x & 31, wv = threadIdx.x >> 5;
  if (lane == 0) red[wv] = ss;
  __syncthreads();
  if (wv == 0) {
    float t = (lane < 8) ? red[lane] : 0.f;
#pragma unroll
    for (int off = 4; off > 0; off >>= 1) t += __shfl_xor(t, off, 32);
    if (lane == 0) red[0] = t;
  }
  __syncthreads();
  const float rinv = rsqrtf(red[0] / (float)HIDDEN + 1e-6f);
  for (int i = threadIdx.x; i < HIDDEN; i += 256)
    xn[(size_t)row * HIDDEN + i] = (half_t)(xr[i] * rinv * w[i]);
}

// ---------------------------------------------------------------------------
// Kernel 2: cast fp32 W[K][N] -> f16 Wt[N][K]  (LDS 32x32 tile transpose)
// grid (N/32, K/32), block (32,8)
// ---------------------------------------------------------------------------
__global__ void cast_transpose(const float* __restrict__ W,
                               half_t* __restrict__ Wt, int K, int N) {
  __shared__ half_t tile[32][33];
  const int n0 = blockIdx.x * 32, k0 = blockIdx.y * 32;
  const int tx = threadIdx.x, ty = threadIdx.y;
#pragma unroll
  for (int i = 0; i < 32; i += 8) {
    int k = k0 + ty + i, n = n0 + tx;
    tile[tx][ty + i] = (half_t)W[(size_t)k * N + n];
  }
  __syncthreads();
#pragma unroll
  for (int i = 0; i < 32; i += 8) {
    int n = n0 + ty + i, k = k0 + tx;
    Wt[(size_t)n * K + k] = tile[ty + i][tx];
  }
}

// ---------------------------------------------------------------------------
// Kernel 3: unified WMMA GEMM.  One wave computes a 32x128 output tile
// (two 16-row sub-tiles x one 128-col group `h`): 2 A frags amortize the
// 8 shared B frags over 16 WMMAs per K step.  One base pointer per stream;
// all sub-tile offsets (n*16*KD = n*64KB, +16*KD for row block 1) are load
// immediates, so the loop carries only two 64-bit addresses.
// launch_bounds(128,1): allow ~200 VGPRs/wave so the 128-VGPR accumulator
// file plus operands fits without spills.
// A: f16 [MTOT][2048] row-major.  Wt: f16 [G*128][2048] (N-major).
// mode 0: fp32 store to outf[row][G*128]          (final projection)
// mode 1: RoPE epilogue, f16 store [b][h][s][128]  (Q, K)
// mode 2: f16 store transposed [b][h][128][s]      (V, d-major for P*V frags)
// grid (MTOT/128, G), block 128 (4 waves, wave w -> 32-row tile blockIdx.x*4+w)
// ---------------------------------------------------------------------------
__global__ void __launch_bounds__(128, 1)
gemm_heads(const half_t* __restrict__ A,
           const half_t* __restrict__ Wt,
           const float* __restrict__ bias,
           const float* __restrict__ cosp,
           const float* __restrict__ sinp,
           half_t* __restrict__ outh,
           float* __restrict__ outf,
           int mode, int G) {
  const int lane   = threadIdx.x & 31;
  const int w      = threadIdx.x >> 5;
  const int mt     = blockIdx.x * 4 + w;
  const int h      = blockIdx.y;
  const int row0   = mt * 32;
  const int col16  = lane & 15;
  const int half8  = (lane < 16) ? 0 : 8;
  const int khalfA = (lane < 16) ? 0 : 16;  // A frag K offset
  const int kbB    = (lane < 16) ? 0 : 16;  // B frag K offset

  v8f acc[2][8];
#pragma unroll
  for (int m = 0; m < 2; ++m)
#pragma unroll
    for (int n = 0; n < 8; ++n)
#pragma unroll
      for (int r = 0; r < 8; ++r) acc[m][n][r] = 0.f;

  // per-lane pre-offset base pointers (the ONLY loop-carried addresses)
  const half_t* pa = A  + (size_t)(row0 + col16) * KD + khalfA;
  const half_t* pb = Wt + (size_t)(h * 128 + col16) * KD + kbB;

  for (int k = 0; k < KD; k += 32) {
    v16h a0 = load_a_frag_pre(pa + k);
    v16h a1 = load_a_frag_pre(pa + 16 * KD + k);
#pragma unroll
    for (int n = 0; n < 8; ++n) {
      v16h b = load_b_frag_pre(pb + k + n * 16 * KD);
      acc[0][n] = wmma_f16(a0, b, acc[0][n]);
      acc[1][n] = wmma_f16(a1, b, acc[1][n]);
    }
  }

  // bias
#pragma unroll
  for (int n = 0; n < 8; ++n) {
    float bv = bias[h * 128 + 16 * n + col16];
#pragma unroll
    for (int m = 0; m < 2; ++m)
#pragma unroll
      for (int r = 0; r < 8; ++r) acc[m][n][r] += bv;
  }

#pragma unroll
  for (int m = 0; m < 2; ++m) {
    const int rbase = row0 + 16 * m;
    if (mode == 0) {                     // fp32 row-major out
#pragma unroll
      for (int n = 0; n < 8; ++n)
#pragma unroll
        for (int r = 0; r < 8; ++r) {
          int row = rbase + half8 + r;
          outf[(size_t)row * (G * 128) + h * 128 + 16 * n + col16] = acc[m][n][r];
        }
    } else if (mode == 1) {              // RoPE + [b][h][s][d] f16
#pragma unroll
      for (int r = 0; r < 8; ++r) {
        int row = rbase + half8 + r;
        int s   = row & (SEQ - 1);
        int b   = row >> 11;             // row / SEQ
        half_t* dst = outh + ((size_t)(b * G + h) * SEQ + s) * HEADDIM;
        const float* cr = cosp + (size_t)s * HEADDIM;
        const float* sr = sinp + (size_t)s * HEADDIM;
#pragma unroll
        for (int n = 0; n < 4; ++n) {
          int d1 = 16 * n + col16, d2 = d1 + 64;
          float lo = acc[m][n][r], hi = acc[m][n + 4][r];
          float o1 = lo * cr[d1] - hi * sr[d1];
          float o2 = hi * cr[d2] + lo * sr[d2];
          dst[d1] = (half_t)o1;
          dst[d2] = (half_t)o2;
        }
      }
    } else {                             // V: [b][h][d][s] f16 (d-major)
#pragma unroll
      for (int n = 0; n < 8; ++n) {
        int d = 16 * n + col16;
#pragma unroll
        for (int r = 0; r < 8; ++r) {
          int row = rbase + half8 + r;
          int s   = row & (SEQ - 1);
          int b   = row >> 11;
          outh[((size_t)(b * G + h) * HEADDIM + d) * SEQ + s] = (half_t)acc[m][n][r];
        }
      }
    }
  }
}

// ---------------------------------------------------------------------------
// Kernel 4: flash attention.  One wave = one (b, h, 16-query-row) tile.
// grid (SEQ/16, NHEADS/4, BATCH), block 128; all 4 waves in a block share the
// same m-tile => identical causal trip counts => __syncthreads is safe.
// ---------------------------------------------------------------------------
__global__ void __launch_bounds__(128, 1)
attn_kernel(const half_t* __restrict__ q,
            const half_t* __restrict__ kk,
            const half_t* __restrict__ vt,
            half_t* __restrict__ ao) {
  __shared__ half_t pl[4][16 * 32];      // per-wave P tile (16 rows x 32 keys)
  const int lane   = threadIdx.x & 31;
  const int w      = threadIdx.x >> 5;
  const int mt     = blockIdx.x;
  const int h      = blockIdx.y * 4 + w;
  const int b      = blockIdx.z;
  const int kvh    = h >> 2;             // GROUP_NUM = 4
  const int s0     = mt * 16;
  const int col16  = lane & 15;
  const int half8  = (lane < 16) ? 0 : 8;
  const int khalfA = (lane < 16) ? 0 : 16;
  const int kbB    = (lane < 16) ? 0 : 16;

  // per-lane pre-offset bases
  const half_t* paq = q  + ((size_t)(b * NHEADS + h) * SEQ + s0 + col16) * HEADDIM + khalfA;
  const half_t* pk  = kk + ((size_t)(b * NKVHEADS + kvh) * SEQ + col16) * HEADDIM + kbB;
  const half_t* pv  = vt + ((size_t)(b * NKVHEADS + kvh) * HEADDIM + col16) * SEQ + kbB;

  v16h qa[4];
#pragma unroll
  for (int t = 0; t < 4; ++t) qa[t] = load_a_frag_pre(paq + 32 * t);

  v8f o[8];
  float mr[8], lr[8];
#pragma unroll
  for (int n = 0; n < 8; ++n)
#pragma unroll
    for (int r = 0; r < 8; ++r) o[n][r] = 0.f;
#pragma unroll
  for (int r = 0; r < 8; ++r) { mr[r] = -1e30f; lr[r] = 0.f; }

  const float scale = 0.08838834764831845f;   // 1/sqrt(128)
  const int jmax = mt >> 1;
  for (int j = 0; j <= jmax; ++j) {
    const int jb = j * 32;

    // scores: 16 queries x 32 keys = 2 frags, 4 WMMA each over d
    v8f sc[2];
#pragma unroll
    for (int t = 0; t < 2; ++t) {
#pragma unroll
      for (int r = 0; r < 8; ++r) sc[t][r] = 0.f;
#pragma unroll
      for (int kblk = 0; kblk < 4; ++kblk) {
        v16h bf = load_b_frag_pre(pk + (size_t)(jb + 16 * t) * HEADDIM + 32 * kblk);
        sc[t] = wmma_f16(qa[kblk], bf, sc[t]);
      }
    }

    // online softmax (per-row state; rows live across 16 lanes)
    float p[2][8];
#pragma unroll
    for (int r = 0; r < 8; ++r) {
      int qs = s0 + half8 + r;
#pragma unroll
      for (int t = 0; t < 2; ++t) {
        int ks = jb + 16 * t + col16;
        float v = sc[t][r] * scale;
        if (ks > qs) v = -1e30f;
        sc[t][r] = v;
      }
      float rowm = fmaxf(sc[0][r], sc[1][r]);
#pragma unroll
      for (int off = 1; off < 16; off <<= 1)
        rowm = fmaxf(rowm, __shfl_xor(rowm, off, 32));
      float mn    = fmaxf(mr[r], rowm);
      float alpha = __expf(mr[r] - mn);
      float psum  = 0.f;
#pragma unroll
      for (int t = 0; t < 2; ++t) {
        float pv2 = __expf(sc[t][r] - mn);
        p[t][r] = pv2;
        psum += pv2;
      }
#pragma unroll
      for (int off = 1; off < 16; off <<= 1) psum += __shfl_xor(psum, off, 32);
      lr[r] = lr[r] * alpha + psum;
      mr[r] = mn;
#pragma unroll
      for (int n = 0; n < 8; ++n) o[n][r] *= alpha;
    }

    // C-layout P -> LDS -> A-layout P
    half_t* pw = pl[w];
#pragma unroll
    for (int r = 0; r < 8; ++r)
#pragma unroll
      for (int t = 0; t < 2; ++t)
        pw[(half8 + r) * 32 + 16 * t + col16] = (half_t)p[t][r];
    __syncthreads();
    v16h pa = load_a_frag_pre(pw + col16 * 32 + khalfA);

    // O += P (16x32) @ V (32x128): 8 WMMA, V pre-transposed d-major
#pragma unroll
    for (int n = 0; n < 8; ++n) {
      v16h bf = load_b_frag_pre(pv + (size_t)(16 * n) * SEQ + jb);
      o[n] = wmma_f16(pa, bf, o[n]);
    }
    __syncthreads();
  }

  // finalize: O / l, store f16 row-major [MTOT][2048]
#pragma unroll
  for (int n = 0; n < 8; ++n)
#pragma unroll
    for (int r = 0; r < 8; ++r) {
      int row = b * SEQ + s0 + half8 + r;
      float ov = o[n][r] / lr[r];
      ao[(size_t)row * HIDDEN + h * 128 + 16 * n + col16] = (half_t)ov;
    }
}

// ---------------------------------------------------------------------------
// Launch.  Workspace layout (f16, 256B aligned), ~80 MB total.
// ---------------------------------------------------------------------------
extern "C" void kernel_launch(void* const* d_in, const int* in_sizes, int n_in,
                              void* d_out, int out_size, void* d_ws, size_t ws_size,
                              hipStream_t stream) {
  const float* x      = (const float*)d_in[0];
  const float* norm_w = (const float*)d_in[1];
  const float* Wq     = (const float*)d_in[2];
  const float* bq     = (const float*)d_in[3];
  const float* Wk     = (const float*)d_in[4];
  const float* bk     = (const float*)d_in[5];
  const float* Wv     = (const float*)d_in[6];
  const float* bv     = (const float*)d_in[7];
  const float* Wo     = (const float*)d_in[8];
  const float* bo     = (const float*)d_in[9];
  const float* cosp   = (const float*)d_in[10];
  const float* sinp   = (const float*)d_in[11];
  float* out          = (float*)d_out;

  char* wsb = (char*)d_ws;
  size_t off = 0;
  auto take = [&](size_t bytes) {
    void* p = wsb + off;
    off = (off + bytes + 255) & ~(size_t)255;
    return p;
  };
  half_t* xn  = (half_t*)take((size_t)MTOT * HIDDEN * 2);
  half_t* wqt = (half_t*)take((size_t)HIDDEN * HIDDEN * 2);
  half_t* wkt = (half_t*)take((size_t)512 * HIDDEN * 2);
  half_t* wvt = (half_t*)take((size_t)512 * HIDDEN * 2);
  half_t* wot = (half_t*)take((size_t)HIDDEN * HIDDEN * 2);
  half_t* qws = (half_t*)take((size_t)BATCH * NHEADS   * SEQ * HEADDIM * 2);
  half_t* kws = (half_t*)take((size_t)BATCH * NKVHEADS * SEQ * HEADDIM * 2);
  half_t* vws = (half_t*)take((size_t)BATCH * NKVHEADS * SEQ * HEADDIM * 2);
  half_t* aws = (half_t*)take((size_t)MTOT * HIDDEN * 2);

  // 1. RMSNorm -> f16
  rmsnorm_cast<<<MTOT, 256, 0, stream>>>(x, norm_w, xn);

  // 2. weights -> f16, transposed (N-major)
  cast_transpose<<<dim3(HIDDEN / 32, HIDDEN / 32), dim3(32, 8), 0, stream>>>(Wq, wqt, HIDDEN, HIDDEN);
  cast_transpose<<<dim3(512 / 32,    HIDDEN / 32), dim3(32, 8), 0, stream>>>(Wk, wkt, HIDDEN, 512);
  cast_transpose<<<dim3(512 / 32,    HIDDEN / 32), dim3(32, 8), 0, stream>>>(Wv, wvt, HIDDEN, 512);
  cast_transpose<<<dim3(HIDDEN / 32, HIDDEN / 32), dim3(32, 8), 0, stream>>>(Wo, wot, HIDDEN, HIDDEN);

  // 3. projections (WMMA, 32-row tiles): Q,K with fused RoPE; V stored d-major
  gemm_heads<<<dim3(MTOT / 128, NHEADS),   128, 0, stream>>>(xn, wqt, bq, cosp, sinp, qws, nullptr, 1, NHEADS);
  gemm_heads<<<dim3(MTOT / 128, NKVHEADS), 128, 0, stream>>>(xn, wkt, bk, cosp, sinp, kws, nullptr, 1, NKVHEADS);
  gemm_heads<<<dim3(MTOT / 128, NKVHEADS), 128, 0, stream>>>(xn, wvt, bv, nullptr, nullptr, vws, nullptr, 2, NKVHEADS);

  // 4. flash attention (WMMA scores + WMMA P*V)
  attn_kernel<<<dim3(SEQ / 16, NHEADS / 4, BATCH), 128, 0, stream>>>(qws, kws, vws, aws);

  // 5. output projection, fp32 result
  gemm_heads<<<dim3(MTOT / 128, NHEADS), 128, 0, stream>>>(aws, wot, bo, nullptr, nullptr, nullptr, out, 0, NHEADS);
}
